// LinearInterpolator_65000035058091
// MI455X (gfx1250) — compile-verified, matrix operand order checked by reading, not run on hardware
//
#include <hip/hip_runtime.h>

typedef __attribute__((ext_vector_type(16))) _Float16 v16h;
typedef __attribute__((ext_vector_type(8)))  float    v8f;

#define DD1 112
#define DD2 224
#define DD3 160
#define NCH 16
#define NVERT 300000
#define VOLS (DD1*DD2*DD3)
#define EPSV 1e-5f

__global__ __launch_bounds__(256) void trilerp_wmma_kernel(
    const float* __restrict__ vert,   // (NVERT, 3)
    const float* __restrict__ vol,    // (16, DD1, DD2, DD3)
    float* __restrict__ out)          // (NVERT, 16)
{
  const int tile = blockIdx.x * (blockDim.x >> 5) + (threadIdx.x >> 5); // 16 verts / wave
  if (tile * 16 >= NVERT) return;                 // wave-uniform: EXEC stays all-1s for WMMA
  const int lane = threadIdx.x & 31;
  const int h = lane >> 4;        // K-half selector for A/B fragments
  const int n = lane & 15;        // channel column (B/D) and row index (A)

  // Every lane computes vertex (tile*16 + n); lanes 16..31 duplicate 0..15 (cache hits).
  const int vtx = tile * 16 + n;
  float px = vert[vtx * 3 + 0];
  float py = vert[vtx * 3 + 1];
  float pz = vert[vtx * 3 + 2];
  px = fminf(fmaxf(px, EPSV), (float)(DD1 - 1) - EPSV);
  py = fminf(fmaxf(py, EPSV), (float)(DD2 - 1) - EPSV);
  pz = fminf(fmaxf(pz, EPSV), (float)(DD3 - 1) - EPSV);
  const float fx = floorf(px), fy = floorf(py), fz = floorf(pz);
  const int x0 = (int)fx, y0 = (int)fy, z0 = (int)fz;
  const float ux = px - fx, uy = py - fy, uz = pz - fz;
  const int baseOff = (x0 * DD2 + y0) * DD3 + z0;

  const float* __restrict__ chan = vol + (size_t)n * VOLS;  // this lane's channel plane

  const float wx0 = 1.0f - ux, wy0 = 1.0f - uy;
  const float w0 = wx0 * wy0;   // q=0: (x0,y0)
  const float w1 = wx0 * uy;    // q=1: (x0,y1)
  const float w2 = ux  * wy0;   // q=2: (x1,y0)
  const float w3 = ux  * uy;    // q=3: (x1,y1)

  v8f acc = {};

  #pragma unroll
  for (int vbase = 0; vbase < 16; vbase += 8) {
    // ---- B fragment: 32x16 f16, B[4v+q, c] = z-lerped corner value --------
    v16h b = {};
    #pragma unroll
    for (int jj = 0; jj < 4; ++jj) {
      // vertex-in-WMMA covered by this lane half: h=0 -> {0,1,4,5}, h=1 -> {2,3,6,7}
      const int vL  = ((jj >> 1) << 2) | (jj & 1) | (h << 1);
      const int src = vbase + vL;                       // lane holding that vertex's data
      const int   bo  = __shfl(baseOff, src, 32);
      const float uzv = __shfl(uz,      src, 32);
      #pragma unroll
      for (int q = 0; q < 4; ++q) {
        const int qx = q >> 1, qy = q & 1;
        const float* pc = chan + bo + qx * (DD2 * DD3) + qy * DD3;
        const float c0 = pc[0];                         // z0
        const float c1 = pc[1];                         // z0+1 (contiguous)
        const float dz = fmaf(uzv, c1 - c0, c0);        // z-lerp in f32
        const int s = ((jj >> 1) << 3) + ((jj & 1) << 2) + q;  // compile-time slot
        b[s] = (_Float16)dz;
      }
    }

    // ---- A fragment: 16x32 f16 block-diagonal weights ----------------------
    v16h a = {};
    #pragma unroll
    for (int s = 0; s < 16; ++s) {
      const int r = s >> 1, p = s & 1;
      const int q  = ((r & 1) << 1) | p;                        // k&3 (h-independent)
      const int vk = ((r >> 2) << 2) + (h << 1) + ((r & 3) >> 1); // k>>2
      const float wq = (q == 0) ? w0 : (q == 1) ? w1 : (q == 2) ? w2 : w3;
      if (n == vk + vbase) a[s] = (_Float16)wq;
    }

    acc = __builtin_amdgcn_wmma_f32_16x16x32_f16(
        /*neg_a=*/false, a, /*neg_b=*/false, b,
        /*c_mod=*/(short)0, acc, /*reuse_a=*/false, /*reuse_b=*/false);
  }

  // ---- Store D: row M = r + 8*h, col = n; 16 lanes -> 64B contiguous per row
  float* __restrict__ orow = out + (size_t)tile * 256;
  #pragma unroll
  for (int r = 0; r < 8; ++r) {
    orow[(r + (h << 3)) * 16 + n] = acc[r];
  }
}

extern "C" void kernel_launch(void* const* d_in, const int* in_sizes, int n_in,
                              void* d_out, int out_size, void* d_ws, size_t ws_size,
                              hipStream_t stream) {
  const float* vert = (const float*)d_in[0];   // (1, 300000, 3) f32
  const float* vol  = (const float*)d_in[1];   // (1, 16, 112, 224, 160) f32
  float* out = (float*)d_out;                  // (1, 300000, 16) f32

  const int tiles = (NVERT + 15) / 16;         // 18750 wave-tiles
  const int wavesPerBlock = 256 / 32;          // 8
  const int blocks = (tiles + wavesPerBlock - 1) / wavesPerBlock;  // 2344

  trilerp_wmma_kernel<<<blocks, 256, 0, stream>>>(vert, vol, out);
}
